// MiMoV2FlashMoE_67894843015332
// MI455X (gfx1250) — matrix operand order repaired and checked by reading, not compiled, not run on hardware
//
#include <hip/hip_runtime.h>
#include <math.h>

// Problem constants (from reference setup_inputs)
#define T_TOK 2048
#define HDIM  1024
#define EDIM  16
#define IDIM  1024
#define PMAX  4608          // 4096 pairs + per-expert padding to 32
#define SUPT  (PMAX / 32)   // 144 supertiles max (M = 32 rows each)
#define LDST1 1032          // mlp1 LDS row stride in bf16 elements (1024 + 8 pad)
#define ROWB2 2112          // mlp2 LDS row stride in bytes (2048 + 2x32B TDM pad)

// Workspace layout (bytes)
#define OFF_COUNTS   0
#define OFF_CURSOR   64
#define OFF_NTILES   128
#define OFF_TILE_E   256
#define OFF_TILE_B   (256 + 4 * SUPT)
#define OFF_TOK2E    4096
#define OFF_TOK2W    (OFF_TOK2E + 4 * T_TOK * 2)
#define OFF_PAIRTOK  (OFF_TOK2W + 4 * T_TOK * 2)
#define OFF_PAIRW    (OFF_PAIRTOK + 4 * PMAX)
#define OFF_ACT      81920

typedef __attribute__((ext_vector_type(16))) __bf16 v16bf;
typedef __attribute__((ext_vector_type(2)))  __bf16 v2bf;
typedef __attribute__((ext_vector_type(8)))  float  v8f;
typedef __attribute__((ext_vector_type(4)))  unsigned int u32x4;
typedef __attribute__((ext_vector_type(8)))  int    i32x8;
typedef __attribute__((ext_vector_type(4)))  int    i32x4;

union BFrag {
  v16bf        bf;
  uint4        q[2];
  unsigned int u[8];
};

// Packed f32x2 -> bf16x2 (hardware v_cvt_pk_bf16_f32 when available)
__device__ __forceinline__ unsigned int pk2(float a, float b) {
#if __has_builtin(__builtin_amdgcn_cvt_pk_bf16_f32)
  v2bf r = __builtin_amdgcn_cvt_pk_bf16_f32(a, b);
  return __builtin_bit_cast(unsigned int, r);
#else
  unsigned int ua = __float_as_uint(a); ua += 0x7FFFu + ((ua >> 16) & 1u);
  unsigned int ub = __float_as_uint(b); ub += 0x7FFFu + ((ub >> 16) & 1u);
  return (ua >> 16) | (ub & 0xFFFF0000u);
#endif
}
__device__ __forceinline__ unsigned short bf1(float v) {
  return (unsigned short)pk2(v, 0.0f);
}

// ---------------------------------------------------------------- init
__global__ __launch_bounds__(256) void k_init(float* __restrict__ out,
                                              int* __restrict__ counts,
                                              int* __restrict__ pairTok,
                                              float* __restrict__ pairW) {
  int i = blockIdx.x * blockDim.x + threadIdx.x;
  int stride = gridDim.x * blockDim.x;
  for (int idx = i; idx < T_TOK * HDIM; idx += stride) out[idx] = 0.0f;
  if (i < PMAX) { pairTok[i] = 0; pairW[i] = 0.0f; }
  if (i < EDIM) counts[i] = 0;
}

// ---------------------------------------------------------------- router
__global__ __launch_bounds__(128) void k_router(const float* __restrict__ x,
                                                const float* __restrict__ rw,
                                                int* __restrict__ counts,
                                                int* __restrict__ tok2e,
                                                float* __restrict__ tok2w) {
  int t = blockIdx.x;
  int tid = threadIdx.x;
  __shared__ float xr[HDIM];
  __shared__ float part[EDIM][8];
  __shared__ float sc[EDIM];

  for (int i = tid; i < HDIM; i += 128) xr[i] = x[(size_t)t * HDIM + i];
  __syncthreads();

  int e = tid & 15, c = tid >> 4;          // 16 experts x 8 chunks
  const float* wrow = rw + (size_t)e * HDIM + c * 128;
  const float* xp = xr + c * 128;
  float s = 0.0f;
  #pragma unroll 8
  for (int h = 0; h < 128; ++h) s += xp[h] * wrow[h];
  part[e][c] = s;
  __syncthreads();

  if (tid < EDIM) {
    float v = 0.0f;
    #pragma unroll
    for (int q = 0; q < 8; ++q) v += part[tid][q];
    sc[tid] = 1.0f / (1.0f + __expf(-v));   // sigmoid
  }
  __syncthreads();

  if (tid == 0) {
    float b1 = -1.0f, b2 = -1.0f; int i1 = 0, i2 = 0;
    #pragma unroll
    for (int q = 0; q < EDIM; ++q) {
      float v = sc[q];
      if (v > b1)      { b2 = b1; i2 = i1; b1 = v; i1 = q; }
      else if (v > b2) { b2 = v; i2 = q; }
    }
    float inv = 1.0f / (b1 + b2 + 1e-20f);
    tok2e[t * 2 + 0] = i1; tok2w[t * 2 + 0] = b1 * inv;
    tok2e[t * 2 + 1] = i2; tok2w[t * 2 + 1] = b2 * inv;
    atomicAdd(&counts[i1], 1);
    atomicAdd(&counts[i2], 1);
  }
}

// ---------------------------------------------------------------- scan (tiny, serial)
__global__ void k_scan(const int* __restrict__ counts, int* __restrict__ cursor,
                       int* __restrict__ tileE, int* __restrict__ tileB,
                       int* __restrict__ nTiles) {
  if (threadIdx.x == 0) {
    int baseSlot = 0, tcnt = 0;
    for (int e = 0; e < EDIM; ++e) {
      int cnt = counts[e];
      int nt = (cnt + 31) >> 5;            // supertiles of 32 pairs
      cursor[e] = baseSlot;
      for (int i = 0; i < nt; ++i) {
        tileE[tcnt] = e;
        tileB[tcnt] = baseSlot + i * 32;
        ++tcnt;
      }
      baseSlot += nt * 32;
    }
    nTiles[0] = tcnt;
  }
}

// ---------------------------------------------------------------- scatter
__global__ __launch_bounds__(256) void k_scatter(const int* __restrict__ tok2e,
                                                 const float* __restrict__ tok2w,
                                                 int* __restrict__ cursor,
                                                 int* __restrict__ pairTok,
                                                 float* __restrict__ pairW) {
  int t = blockIdx.x * blockDim.x + threadIdx.x;
  if (t < T_TOK) {
    #pragma unroll
    for (int k = 0; k < 2; ++k) {
      int e = tok2e[t * 2 + k];
      int slot = atomicAdd(&cursor[e], 1);
      pairTok[slot] = t;
      pairW[slot] = tok2w[t * 2 + k];
    }
  }
}

#define WMMA_BF16(A, B, C) \
  __builtin_amdgcn_wmma_f32_16x16x32_bf16(false, (A), false, (B), (short)0, (C), false, false)

// ---------------------------------------------------------------- gate/up + SwiGLU (phase 1)
// M = 32 pairs per block (2 WMMA sub-tiles share each converted B fragment).
__global__ __launch_bounds__(256) void k_mlp1(
    const float* __restrict__ x, const float* __restrict__ gw_all,
    const float* __restrict__ uw_all, const int* __restrict__ nTiles,
    const int* __restrict__ tileE, const int* __restrict__ tileB,
    const int* __restrict__ pairTok, const float* __restrict__ pairW,
    unsigned short* __restrict__ act) {
  int tile = blockIdx.x;
  if (tile >= nTiles[0]) return;
  int e = tileE[tile];
  int base = tileB[tile];
  int iBase = blockIdx.y * 128;
  int tid = threadIdx.x;

  __shared__ __align__(16) unsigned short xT[32 * LDST1];
  __shared__ float wRow[32];

  // Stage 32 token rows (fp32 -> bf16 via v_cvt_pk) into LDS
  for (int m = 0; m < 32; ++m) {
    int tok = pairTok[base + m];
    float4 f = ((const float4*)(x + (size_t)tok * HDIM))[tid];
    uint2 p; p.x = pk2(f.x, f.y); p.y = pk2(f.z, f.w);
    *(uint2*)&xT[m * LDST1 + tid * 4] = p;
  }
  if (tid < 32) wRow[tid] = pairW[base + tid];
  __syncthreads();

  int lane = tid & 31;
  int wv = tid >> 5;                   // 8 waves -> 8 x 16 columns of I
  int hiHalf = lane >> 4;
  int colN = lane & 15;
  int n = iBase + wv * 16 + colN;

  const float* gp = gw_all + (size_t)e * IDIM * HDIM + (size_t)n * HDIM;
  const float* up = uw_all + (size_t)e * IDIM * HDIM + (size_t)n * HDIM;

  v8f accG0 = {}, accG1 = {}, accU0 = {}, accU1 = {};
  int aOff0 = colN * LDST1;            // sub-tile 0: rows 0..15
  int aOff1 = (16 + colN) * LDST1;     // sub-tile 1: rows 16..31
  int aK0 = hiHalf ? 16 : 0;           // contiguous k = 16h + s (matches B)
  int bK0 = hiHalf ? 16 : 0;           // contiguous k = 16h + s (matches A)

  #pragma unroll 2
  for (int kk = 0; kk < HDIM / 32; ++kk) {
    BFrag A0, A1, Bg, Bu;
    A0.q[0] = *(const uint4*)&xT[aOff0 + kk * 32 + aK0];
    A0.q[1] = *(const uint4*)&xT[aOff0 + kk * 32 + aK0 + 8];
    A1.q[0] = *(const uint4*)&xT[aOff1 + kk * 32 + aK0];
    A1.q[1] = *(const uint4*)&xT[aOff1 + kk * 32 + aK0 + 8];

    const float4* g4 = (const float4*)(gp + kk * 32 + bK0);
    const float4* u4 = (const float4*)(up + kk * 32 + bK0);
    __builtin_prefetch(g4 + 8, 0, 1);     // next k-step of the weight stream
    __builtin_prefetch(u4 + 8, 0, 1);
    float4 g0 = g4[0], g1 = g4[1], g2 = g4[2], g3 = g4[3];
    float4 q0 = u4[0], q1 = u4[1], q2 = u4[2], q3 = u4[3];

    Bg.u[0] = pk2(g0.x, g0.y); Bg.u[1] = pk2(g0.z, g0.w);
    Bg.u[2] = pk2(g1.x, g1.y); Bg.u[3] = pk2(g1.z, g1.w);
    Bg.u[4] = pk2(g2.x, g2.y); Bg.u[5] = pk2(g2.z, g2.w);
    Bg.u[6] = pk2(g3.x, g3.y); Bg.u[7] = pk2(g3.z, g3.w);
    Bu.u[0] = pk2(q0.x, q0.y); Bu.u[1] = pk2(q0.z, q0.w);
    Bu.u[2] = pk2(q1.x, q1.y); Bu.u[3] = pk2(q1.z, q1.w);
    Bu.u[4] = pk2(q2.x, q2.y); Bu.u[5] = pk2(q2.z, q2.w);
    Bu.u[6] = pk2(q3.x, q3.y); Bu.u[7] = pk2(q3.z, q3.w);

    accG0 = WMMA_BF16(A0.bf, Bg.bf, accG0);
    accG1 = WMMA_BF16(A1.bf, Bg.bf, accG1);
    accU0 = WMMA_BF16(A0.bf, Bu.bf, accU0);
    accU1 = WMMA_BF16(A1.bf, Bu.bf, accU1);
  }

  // SwiGLU epilogue, scaled by routing weight; padding rows have w=0 -> exact 0
  #pragma unroll
  for (int j = 0; j < 8; ++j) {
    int r0 = j + 8 * hiHalf;
    float g = accG0[j];
    float v = g / (1.0f + __expf(-g)) * accU0[j] * wRow[r0];
    act[(size_t)(base + r0) * IDIM + n] = bf1(v);
    int r1 = 16 + r0;
    g = accG1[j];
    v = g / (1.0f + __expf(-g)) * accU1[j] * wRow[r1];
    act[(size_t)(base + r1) * IDIM + n] = bf1(v);
  }
}

// ---------------------------------------------------------------- down proj + combine (phase 2)
// Activation tile (32 x 1024 bf16, row stride IDIM) staged to LDS by the
// Tensor Data Mover; LDS rows padded 8 DWORDs per 256 DWORDs (D#.pad) so rows
// land 16 banks apart.
__global__ __launch_bounds__(256) void k_mlp2(
    const unsigned short* __restrict__ act, const float* __restrict__ dw_all,
    const int* __restrict__ nTiles, const int* __restrict__ tileE,
    const int* __restrict__ tileB, const int* __restrict__ pairTok,
    float* __restrict__ out) {
  int tile = blockIdx.x;
  if (tile >= nTiles[0]) return;
  int e = tileE[tile];
  int base = tileB[tile];
  int hBase = blockIdx.y * 128;
  int tid = threadIdx.x;

  __shared__ __align__(16) unsigned char aT[32 * ROWB2];
  __shared__ int tokL[32];

  if (tid < 32) tokL[tid] = pairTok[base + tid];

#if __has_builtin(__builtin_amdgcn_tensor_load_to_lds)
  if (tid < 32) {  // wave 0 issues the TDM DMA for the whole tile
    unsigned long long ga =
        (unsigned long long)(uintptr_t)(act + (size_t)base * IDIM);
    unsigned ldsa = (unsigned)(uintptr_t)(void*)&aT[0];
    unsigned galo = (unsigned)__builtin_amdgcn_readfirstlane((int)(unsigned)(ga & 0xFFFFFFFFull));
    unsigned gahi = (unsigned)__builtin_amdgcn_readfirstlane((int)(unsigned)(ga >> 32));
    ldsa = (unsigned)__builtin_amdgcn_readfirstlane((int)ldsa);

    // D# group 0: count=1 | lds_addr | global_addr | type=2
    u32x4 g0;
    g0[0] = 1u;
    g0[1] = ldsa;
    g0[2] = galo;
    g0[3] = (gahi & 0x01FFFFFFu) | (2u << 30);
    // D# group 1: data_size=2B, pad_enable, pad_interval=256dw, pad_amount=8dw,
    // tensor_dim0=1024, tensor_dim1=32, tile_dim0=1024, tile_dim1=32,
    // tensor_dim0_stride=1024
    i32x8 g1;
    g1[0] = (int)((1u << 16) | (1u << 20) | (7u << 22) | (7u << 25));
    g1[1] = (int)(1024u << 16);   // tensor_dim0[15:0] at bits 63:48
    g1[2] = (int)(32u << 16);     // tensor_dim1[15:0] at bits 95:80
    g1[3] = (int)(1024u << 16);   // tile_dim0 at bits 127:112
    g1[4] = 32;                   // tile_dim1
    g1[5] = 1024;                 // tensor_dim0_stride low 32
    g1[6] = 0;
    g1[7] = 0;
    i32x4 g2; g2[0] = 1; g2[1] = 0; g2[2] = 0; g2[3] = 0;  // tensor_dim2=1 (unused)
    i32x4 g3; g3[0] = 0; g3[1] = 0; g3[2] = 0; g3[3] = 0;
    i32x8 g4; g4[0] = 0; g4[1] = 0; g4[2] = 0; g4[3] = 0;
              g4[4] = 0; g4[5] = 0; g4[6] = 0; g4[7] = 0;  // unused 5th group (zero)
    __builtin_amdgcn_tensor_load_to_lds(g0, g1, g2, g3, g4, 0);
    __builtin_amdgcn_s_wait_tensorcnt((short)0);
  }
#else
  // Manual staging fallback, same padded LDS layout (+32B after each 1024B)
  #pragma unroll
  for (int r = 0; r < 16; ++r) {
    int idx = r * 256 + tid;
    int m = idx >> 7;                 // 128 uint4 per row
    int c = idx & 127;
    uint4 v = ((const uint4*)(act + (size_t)(base + m) * IDIM))[c];
    *(uint4*)&aT[m * ROWB2 + c * 16 + ((c >= 64) ? 32 : 0)] = v;
  }
#endif
  __syncthreads();

  int lane = tid & 31;
  int wv = tid >> 5;
  int hiHalf = lane >> 4;
  int colN = lane & 15;
  int n = hBase + wv * 16 + colN;
  const float* dp = dw_all + (size_t)e * HDIM * IDIM + (size_t)n * IDIM;

  v8f acc0 = {}, acc1 = {};
  int row0 = colN * ROWB2;
  int row1 = (16 + colN) * ROWB2;
  int aK0 = hiHalf ? 16 : 0;
  int bK0 = hiHalf ? 16 : 0;

  #pragma unroll 2
  for (int kk = 0; kk < IDIM / 32; ++kk) {
    int eb = (kk * 32 + aK0) * 2 + ((kk >> 4) << 5);  // +32B past the mid-row pad
    BFrag A0, A1, Bd;
    A0.q[0] = *(const uint4*)&aT[row0 + eb];
    A0.q[1] = *(const uint4*)&aT[row0 + eb + 16];
    A1.q[0] = *(const uint4*)&aT[row1 + eb];
    A1.q[1] = *(const uint4*)&aT[row1 + eb + 16];

    const float4* d4 = (const float4*)(dp + kk * 32 + bK0);
    __builtin_prefetch(d4 + 8, 0, 1);
    float4 d0 = d4[0], d1 = d4[1], d2 = d4[2], d3 = d4[3];
    Bd.u[0] = pk2(d0.x, d0.y); Bd.u[1] = pk2(d0.z, d0.w);
    Bd.u[2] = pk2(d1.x, d1.y); Bd.u[3] = pk2(d1.z, d1.w);
    Bd.u[4] = pk2(d2.x, d2.y); Bd.u[5] = pk2(d2.z, d2.w);
    Bd.u[6] = pk2(d3.x, d3.y); Bd.u[7] = pk2(d3.z, d3.w);

    acc0 = WMMA_BF16(A0.bf, Bd.bf, acc0);
    acc1 = WMMA_BF16(A1.bf, Bd.bf, acc1);
  }

  // Combine: native f32 global atomics (padding rows contribute exact 0.0)
  #pragma unroll
  for (int j = 0; j < 8; ++j) {
    int r0 = j + 8 * hiHalf;
    __hip_atomic_fetch_add(&out[(size_t)tokL[r0] * HDIM + n], acc0[j],
                           __ATOMIC_RELAXED, __HIP_MEMORY_SCOPE_AGENT);
    int r1 = 16 + r0;
    __hip_atomic_fetch_add(&out[(size_t)tokL[r1] * HDIM + n], acc1[j],
                           __ATOMIC_RELAXED, __HIP_MEMORY_SCOPE_AGENT);
  }
}

// ---------------------------------------------------------------- launch
extern "C" void kernel_launch(void* const* d_in, const int* in_sizes, int n_in,
                              void* d_out, int out_size, void* d_ws, size_t ws_size,
                              hipStream_t stream) {
  const float* x  = (const float*)d_in[0];   // [T, H]
  const float* rw = (const float*)d_in[1];   // [E, H]
  const float* gw = (const float*)d_in[2];   // [E, I, H]
  const float* uw = (const float*)d_in[3];   // [E, I, H]
  const float* dw = (const float*)d_in[4];   // [E, H, I]
  float* out = (float*)d_out;

  char* ws = (char*)d_ws;
  int*   counts  = (int*)(ws + OFF_COUNTS);
  int*   cursor  = (int*)(ws + OFF_CURSOR);
  int*   nTiles  = (int*)(ws + OFF_NTILES);
  int*   tileE   = (int*)(ws + OFF_TILE_E);
  int*   tileB   = (int*)(ws + OFF_TILE_B);
  int*   tok2e   = (int*)(ws + OFF_TOK2E);
  float* tok2w   = (float*)(ws + OFF_TOK2W);
  int*   pairTok = (int*)(ws + OFF_PAIRTOK);
  float* pairW   = (float*)(ws + OFF_PAIRW);
  unsigned short* act = (unsigned short*)(ws + OFF_ACT);

  k_init<<<2048, 256, 0, stream>>>(out, counts, pairTok, pairW);
  k_router<<<T_TOK, 128, 0, stream>>>(x, rw, counts, tok2e, tok2w);
  k_scan<<<1, 32, 0, stream>>>(counts, cursor, tileE, tileB, nTiles);
  k_scatter<<<T_TOK / 256, 256, 0, stream>>>(tok2e, tok2w, cursor, pairTok, pairW);
  k_mlp1<<<dim3(SUPT, IDIM / 128), 256, 0, stream>>>(x, gw, uw, nTiles, tileE,
                                                     tileB, pairTok, pairW, act);
  k_mlp2<<<dim3(SUPT, HDIM / 128), 256, 0, stream>>>(act, dw, nTiles, tileE,
                                                     tileB, pairTok, out);
}
